// SequenceHead_1778116461106
// MI455X (gfx1250) — compile-verified, matrix operand order checked by reading, not run on hardware
//
#include <hip/hip_runtime.h>
#include <hip/hip_bf16.h>

typedef __bf16 bf16_t;
typedef __attribute__((ext_vector_type(16))) __bf16 v16bf;
typedef __attribute__((ext_vector_type(8)))  float  v8f;

union FragU { v16bf v; uint4 q[2]; };
union Pack4 { bf16_t h[4]; uint2 u; };

// Load one 16x32 bf16 fragment (A from row-major [rows][K], or B from
// K-transposed weights [cols][K]).  Per CDNA5 ISA 7.12.2, lane l holds
// row/col (l&15); lanes 0-15 pack K = {0..7, 16..23}, lanes 16-31 pack
// K = {8..15, 24..31} -> exactly two 16-byte contiguous loads per lane.
__device__ __forceinline__ v16bf ld_frag(const bf16_t* base, int ldk, int row0, int k0) {
    const int lane = threadIdx.x & 31;
    const bf16_t* p = base + (size_t)(row0 + (lane & 15)) * ldk + k0 + ((lane >> 4) << 3);
    FragU f;
    f.q[0] = *(const uint4*)(p);
    f.q[1] = *(const uint4*)(p + 16);
    return f.v;
}

// One 16x16 f32 output tile of A[Mx K] * B^T[N x K], bias-broadcast init.
__device__ __forceinline__ v8f gemm_tile(const bf16_t* A, const bf16_t* B,
                                         int K, int mt, int nt, float bias) {
    v8f acc = {bias, bias, bias, bias, bias, bias, bias, bias};
    for (int k0 = 0; k0 < K; k0 += 32) {
        v16bf a = ld_frag(A, K, mt * 16, k0);
        v16bf b = ld_frag(B, K, nt * 16, k0);
        acc = __builtin_amdgcn_wmma_f32_16x16x32_bf16(false, a, false, b,
                                                      (short)0, acc, false, false);
    }
    return acc;
}

// ---------------- weight prep: f32 -> bf16, transposed to [N][K] --------------
__global__ void prep_kernel(const float* __restrict__ Wr1, const float* __restrict__ Wr2,
                            const float* __restrict__ Wmsg, const float* __restrict__ Wt1,
                            const float* __restrict__ Wt2, const float* __restrict__ Wo1,
                            const float* __restrict__ Wo2,
                            bf16_t* wr1t, bf16_t* wr2t, bf16_t* wmsgt,
                            bf16_t* wt1t, bf16_t* wt2t, bf16_t* wo1t, bf16_t* wo2t) {
    const int stride = gridDim.x * blockDim.x;
    const int tid = blockIdx.x * blockDim.x + threadIdx.x;
    for (int i = tid; i < 64 * 64; i += stride) {       // [k=64][n=64] -> [n][k]
        int k = i >> 6, n = i & 63;
        wr1t[n * 64 + k] = (bf16_t)Wr1[i];
        wr2t[n * 64 + k] = (bf16_t)Wr2[i];
    }
    for (int i = tid; i < 256 * 128; i += stride) {     // [k=256][n=128] -> [n][k]
        int k = i >> 7, n = i & 127;
        wmsgt[n * 256 + k] = (bf16_t)Wmsg[i];
    }
    for (int i = tid; i < 128 * 128; i += stride) {     // [k=128][n=128] -> [n][k]
        int k = i >> 7, n = i & 127;
        wt1t[n * 128 + k] = (bf16_t)Wt1[i];
        wt2t[n * 128 + k] = (bf16_t)Wt2[i];
        wo1t[n * 128 + k] = (bf16_t)Wo1[i];
    }
    for (int i = tid; i < 32 * 128; i += stride) {      // [k=128][n=20->pad32] -> [n][k]
        int n = i >> 7, k = i & 127;
        wo2t[n * 128 + k] = (n < 20) ? (bf16_t)Wo2[k * 20 + n] : (bf16_t)0.0f;
    }
}

__global__ void zero_kernel(float4* __restrict__ p, int n4) {
    int i = blockIdx.x * blockDim.x + threadIdx.x;
    if (i < n4) p[i] = make_float4(0.f, 0.f, 0.f, 0.f);
}

// ---------------- edge stage: gate MLP + gather/modulate + msg GEMM + scatter -
__global__ __launch_bounds__(128) void edge_kernel(
    const float* __restrict__ atom,   // [N,4,64] = [N,256]
    const float* __restrict__ edgef,  // [E,64]
    const int*   __restrict__ eidx,   // [2,E]
    const float* __restrict__ b_rad1, const float* __restrict__ b_rad2,
    const float* __restrict__ b_msg,
    const bf16_t* __restrict__ Wr1t, const bf16_t* __restrict__ Wr2t,
    const bf16_t* __restrict__ Wmsgt,
    float* __restrict__ scalars, int Etot)
{
    __shared__ __align__(16) bf16_t s_ef[64 * 64];
    __shared__ __align__(16) bf16_t s_hid[64 * 64];
    __shared__ __align__(16) float  s_gate[64 * 64];
    __shared__ __align__(16) bf16_t s_amsg[64 * 256];
    __shared__ int s_src[64], s_dst[64];

    const int tid  = threadIdx.x;
    const int wave = tid >> 5;
    const int lane = tid & 31;
    const int e0   = blockIdx.x * 64;

    // stage edge features (bf16) + indices
    for (int j = tid; j < 64 * 16; j += 128) {
        int e = j >> 4, q = (j & 15) << 2;
        float4 v = *(const float4*)(edgef + (size_t)(e0 + e) * 64 + q);
        Pack4 p;
        p.h[0] = (bf16_t)v.x; p.h[1] = (bf16_t)v.y;
        p.h[2] = (bf16_t)v.z; p.h[3] = (bf16_t)v.w;
        *(uint2*)&s_ef[e * 64 + q] = p.u;
    }
    if (tid < 64) { s_src[tid] = eidx[e0 + tid]; s_dst[tid] = eidx[Etot + e0 + tid]; }
    __syncthreads();

    // GEMM1: silu(ef @ Wr1 + b1) -> s_hid (bf16)   (16 tiles, 4 per wave)
    #pragma unroll
    for (int i = 0; i < 4; ++i) {
        int mt = wave, nt = i;
        int col = nt * 16 + (lane & 15);
        v8f acc = gemm_tile(s_ef, Wr1t, 64, mt, nt, b_rad1[col]);
        int r0 = mt * 16 + ((lane >> 4) << 3);
        #pragma unroll
        for (int r = 0; r < 8; ++r) {
            float x = acc[r];
            x = x / (1.0f + __expf(-x));
            s_hid[(r0 + r) * 64 + col] = (bf16_t)x;
        }
    }
    __syncthreads();

    // GEMM2: hid @ Wr2 + b2 -> gate (f32)
    #pragma unroll
    for (int i = 0; i < 4; ++i) {
        int mt = wave, nt = i;
        int col = nt * 16 + (lane & 15);
        v8f acc = gemm_tile(s_hid, Wr2t, 64, mt, nt, b_rad2[col]);
        int r0 = mt * 16 + ((lane >> 4) << 3);
        #pragma unroll
        for (int r = 0; r < 8; ++r) s_gate[(r0 + r) * 64 + col] = acc[r];
    }
    __syncthreads();

    // gather atom rows (L2-resident), modulate by gate, pack bf16 A-tile
    for (int j = tid; j < 64 * 64; j += 128) {
        int e = j >> 6, q = (j & 63) << 2;   // q in [0,256), channel c = q & 63
        int c = q & 63;
        float4 v = *(const float4*)(atom + (size_t)s_src[e] * 256 + q);
        const float* g = &s_gate[e * 64 + c];
        Pack4 p;
        p.h[0] = (bf16_t)(v.x * g[0]); p.h[1] = (bf16_t)(v.y * g[1]);
        p.h[2] = (bf16_t)(v.z * g[2]); p.h[3] = (bf16_t)(v.w * g[3]);
        *(uint2*)&s_amsg[e * 256 + q] = p.u;
    }
    __syncthreads();

    // GEMM3: amsg[64,256] @ Wmsg[256,128] + b -> scatter atomic add
    {
        int mt = wave;
        #pragma unroll
        for (int nt = 0; nt < 8; ++nt) {
            int col = nt * 16 + (lane & 15);
            v8f acc = gemm_tile(s_amsg, Wmsgt, 256, mt, nt, b_msg[col]);
            int r0 = mt * 16 + ((lane >> 4) << 3);
            #pragma unroll
            for (int r = 0; r < 8; ++r) {
                int e = r0 + r;
                unsafeAtomicAdd(&scalars[(size_t)s_dst[e] * 128 + col], acc[r]);
            }
        }
    }
}

// ---------------- node stage: MLP -> LN -> MLP -> log_softmax -----------------
__global__ __launch_bounds__(128) void node_kernel(
    const float* __restrict__ scalars,
    const float* __restrict__ b_t1, const float* __restrict__ b_t2,
    const float* __restrict__ ln_g, const float* __restrict__ ln_b,
    const float* __restrict__ b_o1, const float* __restrict__ b_o2,
    const bf16_t* __restrict__ Wt1t, const bf16_t* __restrict__ Wt2t,
    const bf16_t* __restrict__ Wo1t, const bf16_t* __restrict__ Wo2t,
    float* __restrict__ out, int Ntot)
{
    __shared__ __align__(16) bf16_t sA[64 * 128];
    __shared__ __align__(16) bf16_t sB[64 * 128];
    __shared__ __align__(16) float  sH[64 * 132];   // padded for bank spread
    __shared__ __align__(16) float  sP[64 * 32];
    const int tid = threadIdx.x, wave = tid >> 5, lane = tid & 31;
    const int n0 = blockIdx.x * 64;

    for (int j = tid; j < 64 * 32; j += 128) {
        int r = j >> 5, q = (j & 31) << 2;
        int n = n0 + r;
        float4 v = make_float4(0.f, 0.f, 0.f, 0.f);
        if (n < Ntot) v = *(const float4*)(scalars + (size_t)n * 128 + q);
        Pack4 p;
        p.h[0] = (bf16_t)v.x; p.h[1] = (bf16_t)v.y;
        p.h[2] = (bf16_t)v.z; p.h[3] = (bf16_t)v.w;
        *(uint2*)&sA[r * 128 + q] = p.u;
    }
    __syncthreads();

    // t1: relu(scalars @ Wt1 + b) -> sB
    #pragma unroll
    for (int nt = 0; nt < 8; ++nt) {
        int mt = wave, col = nt * 16 + (lane & 15);
        v8f acc = gemm_tile(sA, Wt1t, 128, mt, nt, b_t1[col]);
        int r0 = mt * 16 + ((lane >> 4) << 3);
        #pragma unroll
        for (int r = 0; r < 8; ++r)
            sB[(r0 + r) * 128 + col] = (bf16_t)fmaxf(acc[r], 0.0f);
    }
    __syncthreads();

    // t2: sB @ Wt2 + b -> sH (f32)
    #pragma unroll
    for (int nt = 0; nt < 8; ++nt) {
        int mt = wave, col = nt * 16 + (lane & 15);
        v8f acc = gemm_tile(sB, Wt2t, 128, mt, nt, b_t2[col]);
        int r0 = mt * 16 + ((lane >> 4) << 3);
        #pragma unroll
        for (int r = 0; r < 8; ++r) sH[(r0 + r) * 132 + col] = acc[r];
    }
    __syncthreads();

    // LayerNorm per row (f32), renormalized -> sA (bf16)
    if (tid < 64) {
        const float* hr = &sH[tid * 132];
        float mu = 0.f;
        for (int k = 0; k < 128; ++k) mu += hr[k];
        mu *= (1.0f / 128.0f);
        float var = 0.f;
        for (int k = 0; k < 128; ++k) { float d = hr[k] - mu; var += d * d; }
        var *= (1.0f / 128.0f);
        float inv = rsqrtf(var + 1e-5f);
        for (int k = 0; k < 128; ++k)
            sA[tid * 128 + k] = (bf16_t)((hr[k] - mu) * inv * ln_g[k] + ln_b[k]);
    }
    __syncthreads();

    // o1: relu(sA @ Wo1 + b) -> sB
    #pragma unroll
    for (int nt = 0; nt < 8; ++nt) {
        int mt = wave, col = nt * 16 + (lane & 15);
        v8f acc = gemm_tile(sA, Wo1t, 128, mt, nt, b_o1[col]);
        int r0 = mt * 16 + ((lane >> 4) << 3);
        #pragma unroll
        for (int r = 0; r < 8; ++r)
            sB[(r0 + r) * 128 + col] = (bf16_t)fmaxf(acc[r], 0.0f);
    }
    __syncthreads();

    // o2: sB @ Wo2 (N padded to 32) + b -> sP (f32)
    #pragma unroll
    for (int nt = 0; nt < 2; ++nt) {
        int mt = wave, col = nt * 16 + (lane & 15);
        float bias = (col < 20) ? b_o2[col] : 0.0f;
        v8f acc = gemm_tile(sB, Wo2t, 128, mt, nt, bias);
        int r0 = mt * 16 + ((lane >> 4) << 3);
        #pragma unroll
        for (int r = 0; r < 8; ++r) sP[(r0 + r) * 32 + col] = acc[r];
    }
    __syncthreads();

    // log_softmax over 20 classes
    if (tid < 64) {
        int n = n0 + tid;
        if (n < Ntot) {
            const float* pr = &sP[tid * 32];
            float m = pr[0];
            for (int a = 1; a < 20; ++a) m = fmaxf(m, pr[a]);
            float s = 0.f;
            for (int a = 0; a < 20; ++a) s += __expf(pr[a] - m);
            float ls = m + __logf(s);
            for (int a = 0; a < 20; ++a) out[(size_t)n * 20 + a] = pr[a] - ls;
        }
    }
}

extern "C" void kernel_launch(void* const* d_in, const int* in_sizes, int n_in,
                              void* d_out, int out_size, void* d_ws, size_t ws_size,
                              hipStream_t stream) {
    const int Nn = 20000, Ee = 320000;
    const float* atom   = (const float*)d_in[0];
    const float* edgef  = (const float*)d_in[1];
    const int*   eidx   = (const int*)d_in[2];
    const float* W_rad1 = (const float*)d_in[3];
    const float* b_rad1 = (const float*)d_in[4];
    const float* W_rad2 = (const float*)d_in[5];
    const float* b_rad2 = (const float*)d_in[6];
    const float* W_msg  = (const float*)d_in[7];
    const float* b_msg  = (const float*)d_in[8];
    const float* W_t1   = (const float*)d_in[9];
    const float* b_t1   = (const float*)d_in[10];
    const float* W_t2   = (const float*)d_in[11];
    const float* b_t2   = (const float*)d_in[12];
    const float* ln_g   = (const float*)d_in[13];
    const float* ln_b   = (const float*)d_in[14];
    const float* W_o1   = (const float*)d_in[15];
    const float* b_o1   = (const float*)d_in[16];
    const float* W_o2   = (const float*)d_in[17];
    const float* b_o2   = (const float*)d_in[18];
    float* out = (float*)d_out;

    char* ws = (char*)d_ws;
    size_t off = 0;
    float*  scalars = (float*)(ws + off);  off += (size_t)Nn * 128 * 4;  // 10.24 MB
    bf16_t* wr1t  = (bf16_t*)(ws + off);   off += 64 * 64 * 2;
    bf16_t* wr2t  = (bf16_t*)(ws + off);   off += 64 * 64 * 2;
    bf16_t* wmsgt = (bf16_t*)(ws + off);   off += 256 * 128 * 2;
    bf16_t* wt1t  = (bf16_t*)(ws + off);   off += 128 * 128 * 2;
    bf16_t* wt2t  = (bf16_t*)(ws + off);   off += 128 * 128 * 2;
    bf16_t* wo1t  = (bf16_t*)(ws + off);   off += 128 * 128 * 2;
    bf16_t* wo2t  = (bf16_t*)(ws + off);   off += 32 * 128 * 2;

    prep_kernel<<<64, 256, 0, stream>>>(W_rad1, W_rad2, W_msg, W_t1, W_t2, W_o1, W_o2,
                                        wr1t, wr2t, wmsgt, wt1t, wt2t, wo1t, wo2t);
    const int n4 = Nn * 128 / 4;
    zero_kernel<<<(n4 + 255) / 256, 256, 0, stream>>>((float4*)scalars, n4);
    edge_kernel<<<Ee / 64, 128, 0, stream>>>(atom, edgef, eidx, b_rad1, b_rad2, b_msg,
                                             wr1t, wr2t, wmsgt, scalars, Ee);
    node_kernel<<<(Nn + 63) / 64, 128, 0, stream>>>(scalars, b_t1, b_t2, ln_g, ln_b,
                                                    b_o1, b_o2, wt1t, wt2t, wo1t, wo2t,
                                                    out, Nn);
}
